// RGCN_87393994539773
// MI455X (gfx1250) — compile-verified, matrix operand order checked by reading, not run on hardware
//
#include <hip/hip_runtime.h>
#include <hip/hip_bf16.h>
#include <math.h>

#define N_ENT  50000
#define N_REL  24
#define DIM    128
#define N_EDGE 800000
#define BATCH  2048

typedef __attribute__((ext_vector_type(16))) __bf16 v16bf;
typedef __attribute__((ext_vector_type(8)))  __bf16 v8bf;
typedef __attribute__((ext_vector_type(8)))  float  v8f;

// Native FP32 scatter-add: gfx1250 has GLOBAL_ATOMIC_ADD_F32; unsafeAtomicAdd
// pins the native instruction (avoids a CAS retry loop).
__device__ __forceinline__ void atomic_add_f32(float* p, float v) {
    unsafeAtomicAdd(p, v);
}

// ---------------------------------------------------------------------------
// Count edges per (relation, dst) segment.
// ---------------------------------------------------------------------------
__global__ void count_edges_kernel(const int* __restrict__ dst,
                                   const int* __restrict__ etype,
                                   int* __restrict__ cnt) {
    int e = blockIdx.x * blockDim.x + threadIdx.x;
    if (e >= N_EDGE) return;
    atomicAdd(&cnt[etype[e] * N_ENT + dst[e]], 1);
}

// ---------------------------------------------------------------------------
// W_rel (24x128x128) + W_root (128x128): f32 -> bf16, stored TRANSPOSED
// (Wt[mat][n*128+k] = W[mat][k*128+n]) so B-fragment loads are contiguous.
// ---------------------------------------------------------------------------
__global__ void convert_w_kernel(const float* __restrict__ W_rel,
                                 const float* __restrict__ W_root,
                                 __bf16* __restrict__ Wt) {
    int gid = blockIdx.x * blockDim.x + threadIdx.x;
    if (gid >= 25 * DIM * DIM) return;
    int mat = gid / (DIM * DIM);
    int rem = gid - mat * DIM * DIM;
    int k = rem / DIM, n = rem - k * DIM;
    float v = (mat < N_REL) ? W_rel[mat * DIM * DIM + k * DIM + n]
                            : W_root[k * DIM + n];
    Wt[mat * DIM * DIM + n * DIM + k] = (__bf16)v;
}

// ---------------------------------------------------------------------------
// Single-pass scatter into the FULL msg_sum (24 x N x D). One wave per edge.
// ---------------------------------------------------------------------------
__global__ void scatter_all_kernel(const float* __restrict__ x,
                                   const int* __restrict__ src,
                                   const int* __restrict__ dst,
                                   const int* __restrict__ etype,
                                   float* __restrict__ msg) {
    int gid  = blockIdx.x * blockDim.x + threadIdx.x;
    int e    = gid >> 5;
    int lane = gid & 31;
    if (e >= N_EDGE) return;
    int r = etype[e], s = src[e], d = dst[e];
    float*       mrow = msg + ((size_t)r * N_ENT + d) * DIM;
    const float* xrow = x + (size_t)s * DIM;
#pragma unroll
    for (int j = 0; j < 4; ++j) {
        int dim = lane + j * 32;
        atomic_add_f32(&mrow[dim], xrow[dim]);
    }
}

// ---------------------------------------------------------------------------
// Per-relation scatter (fallback path, small workspace).
// ---------------------------------------------------------------------------
__global__ void scatter_kernel(const float* __restrict__ x,
                               const int* __restrict__ src,
                               const int* __restrict__ dst,
                               const int* __restrict__ etype,
                               int rel,
                               float* __restrict__ msg) {
    int gid  = blockIdx.x * blockDim.x + threadIdx.x;
    int e    = gid >> 5;
    int lane = gid & 31;
    if (e >= N_EDGE) return;
    if (etype[e] != rel) return;
    int s = src[e], d = dst[e];
#pragma unroll
    for (int j = 0; j < 4; ++j) {
        int dim = lane + j * 32;
        atomic_add_f32(&msg[d * DIM + dim], x[s * DIM + dim]);
    }
}

// ---------------------------------------------------------------------------
// Shared WMMA micro-kernel body: one wave computes a 16x16 f32 tile over
// K=128 as four 16x16x32 bf16 WMMAs. A comes from LDS (ISA 16-bit A layout:
// lane l holds row l&15, K bytes [h*8 .. +7] and [16+h*8 .. +7] per chunk).
// ---------------------------------------------------------------------------
__device__ __forceinline__ v8f wmma_k128(const __bf16* As, const __bf16* Bt,
                                         int m, int h, int ncol, v8f acc) {
#pragma unroll
    for (int kb = 0; kb < 4; ++kb) {
        const int base1 = kb * 32 + h * 8;
        const int base2 = base1 + 16;
        v8bf alo = *(const v8bf*)(&As[m * DIM + base1]);
        v8bf ahi = *(const v8bf*)(&As[m * DIM + base2]);
        v8bf blo = *(const v8bf*)(&Bt[ncol * DIM + base1]);
        v8bf bhi = *(const v8bf*)(&Bt[ncol * DIM + base2]);
        v16bf a, b;
#pragma unroll
        for (int j = 0; j < 8; ++j) {
            a[j] = alo[j]; a[8 + j] = ahi[j];
            b[j] = blo[j]; b[8 + j] = bhi[j];
        }
        acc = __builtin_amdgcn_wmma_f32_16x16x32_bf16(
                  false, a, false, b, (short)0, acc, false, false);
    }
    return acc;
}

// ---------------------------------------------------------------------------
// FUSED GEMM (big-workspace path): each block owns a 16-node tile and loops
// over all 25 matrices (24 relations + root), accumulating 16x128 entirely
// in registers. agg written exactly once with plain stores.
// ---------------------------------------------------------------------------
__global__ void __launch_bounds__(256)
rgcn_fused_gemm_kernel(const float* __restrict__ msg,   // 24 x N x D (f32)
                       const float* __restrict__ x,     // N x D
                       const int*   __restrict__ cnt,   // 24 x N
                       const __bf16* __restrict__ Wt,   // 25 x D x D, transposed
                       float* __restrict__ agg) {
    __shared__ __bf16 As[16 * DIM];                     // 4 KB

    const int t    = threadIdx.x;
    const int tile = blockIdx.x;
    const int row  = t >> 4;
    const int col0 = (t & 15) * 8;
    const int node = tile * 16 + row;
    const int wave = t >> 5;
    const int lane = t & 31;
    const int m    = lane & 15;
    const int h    = lane >> 4;
    const int ncol = wave * 16 + m;

    v8f acc = {};
    for (int mat = 0; mat < N_REL + 1; ++mat) {
        const float* Asrc;
        float scale;
        if (mat < N_REL) {                              // uniform branch
            Asrc = msg + (size_t)mat * N_ENT * DIM;
            int c = cnt[mat * N_ENT + node];
            scale = 1.0f / (float)(c > 1 ? c : 1);
        } else {
            Asrc = x;
            scale = 1.0f;
        }
        __syncthreads();                                // As safe to overwrite
#pragma unroll
        for (int j = 0; j < 8; ++j)
            As[row * DIM + col0 + j] =
                (__bf16)(Asrc[(size_t)node * DIM + col0 + j] * scale);
        __syncthreads();
        acc = wmma_k128(As, Wt + mat * DIM * DIM, m, h, ncol, acc);
    }

    // C/D layout: VGPR v holds M = v + 8*h, N = lane&15
#pragma unroll
    for (int v = 0; v < 8; ++v)
        agg[(tile * 16 + h * 8 + v) * DIM + ncol] = acc[v];
}

// ---------------------------------------------------------------------------
// Per-relation GEMM (fallback path). ACCUM templated -> branch-free writeback.
// ---------------------------------------------------------------------------
template <bool ACCUM>
__global__ void __launch_bounds__(256)
gemm16_kernel(const float* __restrict__ A,
              const int* __restrict__ cnt,              // may be null
              const __bf16* __restrict__ Bt,
              float* __restrict__ agg) {
    __shared__ __bf16 As[16 * DIM];

    const int t    = threadIdx.x;
    const int tile = blockIdx.x;
    {
        int row  = t >> 4;
        int col0 = (t & 15) * 8;
        int node = tile * 16 + row;
        float scale = 1.0f;
        if (cnt) {
            int c = cnt[node];
            scale = 1.0f / (float)(c > 1 ? c : 1);
        }
#pragma unroll
        for (int j = 0; j < 8; ++j)
            As[row * DIM + col0 + j] = (__bf16)(A[node * DIM + col0 + j] * scale);
    }
    __syncthreads();

    const int wave = t >> 5;
    const int lane = t & 31;
    const int m    = lane & 15;
    const int h    = lane >> 4;
    const int ncol = wave * 16 + m;

    v8f acc = {};
    acc = wmma_k128(As, Bt, m, h, ncol, acc);

#pragma unroll
    for (int v = 0; v < 8; ++v) {
        int idx = (tile * 16 + h * 8 + v) * DIM + ncol;
        if (ACCUM) agg[idx] += acc[v];
        else       agg[idx]  = acc[v];
    }
}

// ---------------------------------------------------------------------------
// out[b,:] = tanh(agg[h]) * init_rel[r] * tanh(agg[t])
// ---------------------------------------------------------------------------
__global__ void gather_kernel(const float* __restrict__ agg,
                              const float* __restrict__ init_rel,
                              const int* __restrict__ sample,
                              float* __restrict__ out) {
    int b = blockIdx.x;
    int d = threadIdx.x;
    int hidx = sample[b * 3 + 0];
    int ridx = sample[b * 3 + 1];
    int tidx = sample[b * 3 + 2];
    out[b * DIM + d] = tanhf(agg[hidx * DIM + d]) *
                       init_rel[ridx * DIM + d] *
                       tanhf(agg[tidx * DIM + d]);
}

// ---------------------------------------------------------------------------
extern "C" void kernel_launch(void* const* d_in, const int* in_sizes, int n_in,
                              void* d_out, int out_size, void* d_ws, size_t ws_size,
                              hipStream_t stream) {
    const float* init_embed = (const float*)d_in[0];
    const float* init_rel   = (const float*)d_in[1];
    const float* W_rel      = (const float*)d_in[2];
    const float* W_root     = (const float*)d_in[3];
    const int*   edge_index = (const int*)d_in[4];
    const int*   edge_type  = (const int*)d_in[5];
    const int*   sample     = (const int*)d_in[6];
    float*       out        = (float*)d_out;

    // workspace layout: cnt | agg | Wt | msg (msg sized per path)
    char* ws = (char*)d_ws;
    const size_t cnt_bytes  = (size_t)N_REL * N_ENT * sizeof(int);        //  4.8 MB
    const size_t agg_bytes  = (size_t)N_ENT * DIM * sizeof(float);        // 25.6 MB
    const size_t wt_bytes   = (size_t)25 * DIM * DIM * sizeof(__bf16);    //  0.8 MB
    const size_t msg1_bytes = (size_t)N_ENT * DIM * sizeof(float);        // 25.6 MB
    const size_t msgF_bytes = (size_t)N_REL * N_ENT * DIM * sizeof(float);// 614.4 MB

    int*    cnt = (int*)ws;
    float*  agg = (float*)(ws + cnt_bytes);
    __bf16* Wt  = (__bf16*)(ws + cnt_bytes + agg_bytes);
    float*  msg = (float*)(ws + cnt_bytes + agg_bytes + wt_bytes);

    const int* esrc = edge_index;            // edge_index[0, :]
    const int* edst = edge_index + N_EDGE;   // edge_index[1, :]

    const bool big_ws =
        ws_size >= cnt_bytes + agg_bytes + wt_bytes + msgF_bytes;

    hipMemsetAsync(cnt, 0, cnt_bytes, stream);
    count_edges_kernel<<<(N_EDGE + 255) / 256, 256, 0, stream>>>(edst, edge_type, cnt);
    convert_w_kernel<<<(25 * DIM * DIM + 255) / 256, 256, 0, stream>>>(W_rel, W_root, Wt);

    if (big_ws) {
        // --- optimal path: one scatter pass + one fused register-resident GEMM
        hipMemsetAsync(msg, 0, msgF_bytes, stream);
        scatter_all_kernel<<<(N_EDGE * 32) / 256, 256, 0, stream>>>(
            init_embed, esrc, edst, edge_type, msg);
        rgcn_fused_gemm_kernel<<<N_ENT / 16, 256, 0, stream>>>(
            msg, init_embed, cnt, Wt, agg);
    } else {
        // --- fallback: per-relation chunking, agg accumulated in HBM
        gemm16_kernel<false><<<N_ENT / 16, 256, 0, stream>>>(
            init_embed, nullptr, Wt + 24 * DIM * DIM, agg);
        for (int r = 0; r < N_REL; ++r) {
            hipMemsetAsync(msg, 0, msg1_bytes, stream);
            scatter_kernel<<<(N_EDGE * 32) / 256, 256, 0, stream>>>(
                init_embed, esrc, edst, edge_type, r, msg);
            gemm16_kernel<true><<<N_ENT / 16, 256, 0, stream>>>(
                msg, cnt + r * N_ENT, Wt + r * DIM * DIM, agg);
        }
    }

    gather_kernel<<<BATCH, DIM, 0, stream>>>(agg, init_rel, sample, out);
}
